// CausalSelfAttention_17188459119385
// MI455X (gfx1250) — matrix operand order repaired and checked by reading, not compiled, not run on hardware
//
#include <hip/hip_runtime.h>
#include <hip/hip_bf16.h>

typedef __attribute__((ext_vector_type(16))) __bf16 v16bf;
typedef __attribute__((ext_vector_type(8)))  float  v8f;

union F16 { v16bf v; unsigned u[8]; __bf16 h[16]; };

__device__ __forceinline__ v8f wmma_bf16(v16bf a, v16bf b, v8f c) {
  // D = A(16x32 bf16) * B(32x16 bf16) + C(16x16 f32)
  return __builtin_amdgcn_wmma_f32_16x16x32_bf16(false, a, false, b, (short)0, c, false, false);
}

__device__ __forceinline__ unsigned pack2(float a, float b) {
  union { __bf16 h[2]; unsigned u; } p;
  p.h[0] = (__bf16)a; p.h[1] = (__bf16)b;
  return p.u;
}

// DPP16 row-rotate move (stays in VALU pipe; row = 16 lanes on gfx1250).
template <int CTRL>
__device__ __forceinline__ float dpp_rot(float x) {
  const int xi = __builtin_bit_cast(int, x);
  const int r  = __builtin_amdgcn_update_dpp(xi, xi, CTRL, 0xf, 0xf, true);
  return __builtin_bit_cast(float, r);
}
// Rotate-reduce over the 16-lane group: result broadcast to all 16 lanes.
__device__ __forceinline__ float red16_max(float x) {
  x = fmaxf(x, dpp_rot<0x128>(x));  // row_ror:8
  x = fmaxf(x, dpp_rot<0x124>(x));  // row_ror:4
  x = fmaxf(x, dpp_rot<0x122>(x));  // row_ror:2
  x = fmaxf(x, dpp_rot<0x121>(x));  // row_ror:1
  return x;
}
__device__ __forceinline__ float red16_sum(float x) {
  x += dpp_rot<0x128>(x);
  x += dpp_rot<0x124>(x);
  x += dpp_rot<0x122>(x);
  x += dpp_rot<0x121>(x);
  return x;
}

// ---------------------------------------------------------------------------
// Generic GEMM:  C[M,N] = A[M,K] * W[K,N] + bias[N]
// A is f32 or bf16 (template), W/bias f32, C f32 or bf16 (template).
// Block tile 128x64, K-step 32, 8 waves: wave w owns rows [16w,16w+16) x 64 cols.
// ---------------------------------------------------------------------------
template <bool A_F32, bool O_F32>
__global__ __launch_bounds__(256) void gemm_wmma_kernel(
    const void* __restrict__ Ap, int lda,
    const float* __restrict__ W, int ldw,
    const float* __restrict__ bias,
    void* __restrict__ Cp, int ldc, int Ktot)
{
  __shared__ __bf16 As[128 * 34];   // [row][k], stride 34 (pad 2)
  __shared__ __bf16 Bs[64 * 34];    // [n][k],  stride 34 (pad 2)

  const int tid  = threadIdx.x;
  const int lane = tid & 31;
  const int wid  = tid >> 5;
  const int rbase = wid * 16;
  const int rowBlk = blockIdx.y * 128;
  const int nbase  = blockIdx.x * 64;

  const int r    = lane & 15;
  const int nloc = lane & 15;
  const int kbA  = (lane < 16) ? 0 : 8;    // A-frag K base (16-bit layout)
  const int kbB  = (lane < 16) ? 0 : 8;    // B-frag K base
  const int m0   = (lane < 16) ? 0 : 8;    // C-frag row base

  v8f acc[4];
#pragma unroll
  for (int j = 0; j < 4; ++j) acc[j] = v8f{};

  for (int kt = 0; kt < Ktot; kt += 32) {
    // ---- stage A tile (128 x 32) as bf16 pairs ----
    for (int idx = tid; idx < 2048; idx += 256) {
      const int row = idx >> 4, pc = idx & 15;
      const int g = (rowBlk + row) * lda + kt + pc * 2;
      unsigned pk;
      if constexpr (A_F32) {
        const float2 v = *(const float2*)((const float*)Ap + g);
        pk = pack2(v.x, v.y);
      } else {
        pk = *(const unsigned*)((const __bf16*)Ap + g);
      }
      *(unsigned*)(As + row * 34 + pc * 2) = pk;
    }
    // ---- stage W tile (32 x 64), transposed to [n][k] bf16 ----
    for (int idx = tid; idx < 1024; idx += 256) {
      const int pc = idx & 31, k = idx >> 5;
      const float2 w = *(const float2*)(W + (kt + k) * ldw + nbase + pc * 2);
      Bs[(2 * pc)     * 34 + k] = (__bf16)w.x;
      Bs[(2 * pc + 1) * 34 + k] = (__bf16)w.y;
    }
    // ---- prefetch next K-step's tiles into cache (global_prefetch_b8) ----
    if (kt + 32 < Ktot) {
      const int prow = tid >> 4, ppc = tid & 15;
      if constexpr (A_F32)
        __builtin_prefetch((const float*)Ap + (rowBlk + prow) * lda + kt + 32 + ppc * 2, 0, 0);
      else
        __builtin_prefetch((const __bf16*)Ap + (rowBlk + prow) * lda + kt + 32 + ppc * 2, 0, 0);
      __builtin_prefetch(W + (kt + 32 + (tid >> 5)) * ldw + nbase + (tid & 31) * 2, 0, 0);
    }
    __syncthreads();

    // ---- A fragment (16x32, ISA 16-bit layout) ----
    F16 fa;
    {
      const __bf16* base = As + (rbase + r) * 34;
#pragma unroll
      for (int i = 0; i < 8; ++i) {
        const int k = (i < 4) ? (kbA + 2 * i) : (16 + kbA + 2 * (i - 4));
        fa.u[i] = *(const unsigned*)(base + k);
      }
    }
    // ---- 4 output tiles along N ----
#pragma unroll
    for (int j = 0; j < 4; ++j) {
      F16 fb;
      const __bf16* bb = Bs + (j * 16 + nloc) * 34 + kbB;
#pragma unroll
      for (int i = 0; i < 8; ++i) fb.u[i] = *(const unsigned*)(bb + 2 * i + ((i < 4) ? 0 : 8));
      acc[j] = wmma_bf16(fa.v, fb.v, acc[j]);
    }
    __syncthreads();
  }

  // ---- epilogue: bias + store (C layout: lane holds col nloc, rows m0..m0+7) ----
#pragma unroll
  for (int j = 0; j < 4; ++j) {
    const int col = nbase + j * 16 + nloc;
    const float bi = bias[col];
#pragma unroll
    for (int i = 0; i < 8; ++i) {
      const int row = rowBlk + rbase + m0 + i;
      const float v = acc[j][i] + bi;
      if constexpr (O_F32) ((float*)Cp)[row * ldc + col] = v;
      else                 ((__bf16*)Cp)[row * ldc + col] = (__bf16)v;
    }
  }
}

// ---------------------------------------------------------------------------
// Flash attention (causal, GQA). One workgroup = 128 query rows of one head.
// 8 waves x 16-row query tiles; K/V streamed in 32-key chunks, double-buffered
// in LDS via GLOBAL_LOAD_ASYNC_TO_LDS_B128 (ASYNCcnt) to overlap fetch w/ WMMA.
// Chunks fully below the diagonal skip masking; chunks fully above a wave's
// tile skip all compute (wave-uniform branches; barriers stay uniform).
// ---------------------------------------------------------------------------
__global__ __launch_bounds__(256) void attn_wmma_kernel(
    const __bf16* __restrict__ Q, const __bf16* __restrict__ K,
    const __bf16* __restrict__ V, __bf16* __restrict__ O)
{
  constexpr int T = 2048;
  constexpr int KS = 72;                 // K/V tile stride: 144B rows (16B aligned)
  __shared__ __bf16 Kt[2][32 * KS];      // [buf][s][d]
  __shared__ __bf16 Vt[2][32 * KS];
  __shared__ __bf16 Ps[8 * 16 * 34];     // per-wave P tile [m][s_local]

  const int tid  = threadIdx.x;
  const int lane = tid & 31;
  const int wid  = tid >> 5;
  const int bh = blockIdx.y;
  const int b  = bh >> 4;
  const int h  = bh & 15;
  const int kv = h >> 2;                 // head h = kv*4 + g
  const int qbase = blockIdx.x * 128;

  const int r    = lane & 15;
  const int nloc = lane & 15;
  const int kbA  = (lane < 16) ? 0 : 8;
  const int kbB  = (lane < 16) ? 0 : 8;
  const int m0   = (lane < 16) ? 0 : 8;

  // Each thread owns one 16-byte chunk of the 32x64 bf16 K (and V) tile.
  const int sA = tid >> 3;               // key row 0..31
  const int cA = (tid & 7) * 8;          // bf16 column 0..56 step 8 (16B)

  auto issue_async = [&](int s0n, int buf) {
    const int koff = (b * T + s0n + sA) * 256 + kv * 64 + cA;
    const unsigned lk = (unsigned)(size_t)(&Kt[buf][sA * KS + cA]);
    const unsigned lv = (unsigned)(size_t)(&Vt[buf][sA * KS + cA]);
    asm volatile(
        "global_load_async_to_lds_b128 %0, %1, off\n\t"
        "global_load_async_to_lds_b128 %2, %3, off"
        :: "v"(lk), "v"((unsigned long long)(size_t)(K + koff)),
           "v"(lv), "v"((unsigned long long)(size_t)(V + koff))
        : "memory");
  };

  // ---- Q fragments for this wave's 16 rows (two K-steps over D=64) ----
  const int qt0  = qbase + wid * 16;     // first query row of this wave's tile
  const int qrow = qt0 + r;
  const int qoff = (b * T + qrow) * 1024 + h * 64;
  F16 qf[2];
#pragma unroll
  for (int ks = 0; ks < 2; ++ks)
#pragma unroll
    for (int i = 0; i < 8; ++i) {
      const int d = ks * 32 + ((i < 4) ? (kbA + 2 * i) : (16 + kbA + 2 * (i - 4)));
      qf[ks].u[i] = *(const unsigned*)(Q + qoff + d);
    }

  v8f oacc[4];
#pragma unroll
  for (int j = 0; j < 4; ++j) oacc[j] = v8f{};
  float mrow[8], lrow[8];
#pragma unroll
  for (int i = 0; i < 8; ++i) { mrow[i] = -3.0e38f; lrow[i] = 0.0f; }

  __bf16* Pw = Ps + wid * 16 * 34;
  const int send = qbase + 128;

  issue_async(0, 0);                     // prime the pipeline
  int cur = 0;

  for (int s0 = 0; s0 < send; s0 += 32) {
    asm volatile("s_wait_asynccnt 0x0" ::: "memory");  // our chunk has landed
    __syncthreads();                     // all waves done reading buf cur^1
    if (s0 + 32 < send) issue_async(s0 + 32, cur ^ 1); // prefetch next chunk

    const __bf16* Ktc = Kt[cur];
    const __bf16* Vtc = Vt[cur];

    const bool live     = (s0 <= qt0 + 15);       // chunk intersects causal region
    const bool needMask = (s0 + 31 > qt0);        // chunk crosses the diagonal
    float al[8];

    if (live) {
      // ---- scores: two 16x16 tiles over the 32-key chunk ----
      v8f sc[2];
#pragma unroll
      for (int t = 0; t < 2; ++t) {
        F16 kf0, kf1;  // B = K^T: lane col = key, K-dim = d
        const __bf16* kb_ = Ktc + (t * 16 + nloc) * KS + kbB;
#pragma unroll
        for (int i = 0; i < 8; ++i) {
          kf0.u[i] = *(const unsigned*)(kb_ + 2 * i + ((i < 4) ? 0 : 8));         // d in [0,32)
          kf1.u[i] = *(const unsigned*)(kb_ + 32 + 2 * i + ((i < 4) ? 0 : 8));    // d in [32,64)
        }
        v8f c = v8f{};
        c = wmma_bf16(qf[0].v, kf0.v, c);
        c = wmma_bf16(qf[1].v, kf1.v, c);
#pragma unroll
        for (int i = 0; i < 8; ++i) sc[t][i] = c[i] * 0.125f;  // 1/sqrt(64)
        if (needMask) {
          const int sg = s0 + t * 16 + nloc;
#pragma unroll
          for (int i = 0; i < 8; ++i)
            if (sg > qt0 + m0 + i) sc[t][i] = -3.0e38f;
        }
      }

      // ---- online softmax (row stats via DPP16 rotate-reductions) ----
#pragma unroll
      for (int i = 0; i < 8; ++i) {
        const float mx = red16_max(fmaxf(sc[0][i], sc[1][i]));
        const float mn = fmaxf(mrow[i], mx);
        al[i] = __expf(mrow[i] - mn);
        mrow[i] = mn;
        const float p0 = __expf(sc[0][i] - mn);
        const float p1 = __expf(sc[1][i] - mn);
        Pw[(m0 + i) * 34 + nloc]      = (__bf16)p0;
        Pw[(m0 + i) * 34 + 16 + nloc] = (__bf16)p1;
        lrow[i] = lrow[i] * al[i] + red16_sum(p0 + p1);
      }
    }
    __syncthreads();  // make P visible to all lanes (uniform across waves)

    if (live) {
      // ---- P fragment (16x32 A-layout) ----
      F16 pf;
      const __bf16* pb = Pw + r * 34;
#pragma unroll
      for (int i = 0; i < 8; ++i) {
        const int k = (i < 4) ? (kbA + 2 * i) : (16 + kbA + 2 * (i - 4));
        pf.u[i] = *(const unsigned*)(pb + k);
      }
      // ---- rescale accumulators and O += P * V ----
#pragma unroll
      for (int j = 0; j < 4; ++j) {
#pragma unroll
        for (int i = 0; i < 8; ++i) oacc[j][i] *= al[i];
        F16 vf;  // B = V: lane col = d, K-dim = s
#pragma unroll
        for (int i = 0; i < 8; ++i) {
          vf.h[2 * i]     = Vtc[(kbB + 2 * i + ((i < 4) ? 0 : 8))     * KS + j * 16 + nloc];
          vf.h[2 * i + 1] = Vtc[(kbB + 2 * i + 1 + ((i < 4) ? 0 : 8)) * KS + j * 16 + nloc];
        }
        oacc[j] = wmma_bf16(pf.v, vf.v, oacc[j]);
      }
    }
    cur ^= 1;
  }

  // ---- normalize and store (bf16, head-major channels) ----
#pragma unroll
  for (int i = 0; i < 8; ++i) {
    const float inv = 1.0f / lrow[i];
    const int row = qt0 + m0 + i;
    const int ob = (b * T + row) * 1024 + h * 64;
#pragma unroll
    for (int j = 0; j < 4; ++j)
      O[ob + j * 16 + nloc] = (__bf16)(oacc[j][i] * inv);
  }
}

// ---------------------------------------------------------------------------
extern "C" void kernel_launch(void* const* d_in, const int* in_sizes, int n_in,
                              void* d_out, int out_size, void* d_ws, size_t ws_size,
                              hipStream_t stream) {
  const float* x   = (const float*)d_in[0];
  const float* w_q = (const float*)d_in[1];
  const float* b_q = (const float*)d_in[2];
  const float* w_k = (const float*)d_in[3];
  const float* b_k = (const float*)d_in[4];
  const float* w_v = (const float*)d_in[5];
  const float* b_v = (const float*)d_in[6];
  const float* w_o = (const float*)d_in[7];
  const float* b_o = (const float*)d_in[8];

  constexpr int BT = 2 * 2048;   // 4096 rows
  __bf16* Qws = (__bf16*)d_ws;                 // [4096,1024]
  __bf16* Kws = Qws + (size_t)BT * 1024;       // [4096,256]
  __bf16* Vws = Kws + (size_t)BT * 256;        // [4096,256]
  __bf16* Aws = Vws + (size_t)BT * 256;        // [4096,1024]

  dim3 blk(256);
  // Q/K/V projections (f32 in -> bf16 out)
  gemm_wmma_kernel<true, false><<<dim3(1024 / 64, BT / 128), blk, 0, stream>>>(
      x, 1024, w_q, 1024, b_q, Qws, 1024, 1024);
  gemm_wmma_kernel<true, false><<<dim3(256 / 64, BT / 128), blk, 0, stream>>>(
      x, 1024, w_k, 256, b_k, Kws, 256, 1024);
  gemm_wmma_kernel<true, false><<<dim3(256 / 64, BT / 128), blk, 0, stream>>>(
      x, 1024, w_v, 256, b_v, Vws, 256, 1024);
  // Causal GQA attention
  attn_wmma_kernel<<<dim3(2048 / 128, 2 * 16), blk, 0, stream>>>(Qws, Kws, Vws, Aws);
  // Output projection (bf16 in -> f32 out)
  gemm_wmma_kernel<false, true><<<dim3(1024 / 64, BT / 128), blk, 0, stream>>>(
      Aws, 1024, w_o, 1024, b_o, (float*)d_out, 1024, 1024);
}
